// Network_81458349735974
// MI455X (gfx1250) — compile-verified, matrix operand order checked by reading, not run on hardware
//
#include <hip/hip_runtime.h>

// ---------------------------------------------------------------------------
// CDNA5 (gfx1250) implementation of the SHS-Net style point network.
// All 1x1 convs run as batched GEMMs on the f16 WMMA path
// (v_wmma_f32_16x16x32_f16, f32 accumulate), with pre-converted, zero-padded
// f16 operands (weights row-major, activations transposed to n-major) so the
// WMMA inner loop is pure aligned b128 loads + wmma (no guards, no cvt).
// Each wave runs 4 independent accumulator chains (16x64 output) for A-reuse
// and WMMA hazard covering. BN is two-pass (stats + apply). Edge convs
// recompute instead of materializing (B,24,N,K).
// ---------------------------------------------------------------------------

typedef __attribute__((ext_vector_type(16))) _Float16 v16h;
typedef __attribute__((ext_vector_type(8)))  _Float16 v8h;
typedef __attribute__((ext_vector_type(8)))  float    v8f;

#define BB 128
#define NPAT 700
#define NSAM 1200
#define EDGE_O 24

static inline long cdivl(long a, long b) { return (a + b - 1) / b; }
static inline int ALN16(int n) { return (n + 15) & ~15; }
static inline int ALN32(int n) { return (n + 31) & ~31; }

// ---------------------------------------------------------------------------
// Param index tables (pytree flatten: dicts with sorted keys).
// Top level: net (36), pe (56), pg (56); inputs: [pcl_pat, pcl_sample, params].
// ---------------------------------------------------------------------------
enum { // pe / pg block, alphabetical
  P_ALPHA = 0, P_BETA,
  P_C1_B, P_C1_BE, P_C1_G, P_C1_W,
  P_C2_B, P_C2_BE, P_C2_G, P_C2_W,
  P_C3_B, P_C3_BE, P_C3_G, P_C3_W,
  P_G0_B, P_G0_BE, P_G0_G, P_G0_W,
  P_G1_B, P_G1_BE, P_G1_G, P_G1_W,
  P_LIN0_B, P_LIN0_W, P_LIN1_B, P_LIN1_W, P_LIN2_B, P_LIN2_W,
  P_LIN3_B, P_LIN3_W, P_LIN4_B, P_LIN4_W,
  P_M10_B, P_M10_W, P_M11_B, P_M11_W,
  P_P1_B, P_P1_BE, P_P1_G, P_P1_W,
  P_P2_B, P_P2_BE, P_P2_G, P_P2_W,
  P_P3_B, P_P3_BE, P_P3_G, P_P3_W,
  P_P4_B, P_P4_BE, P_P4_G, P_P4_W,
  P_P5_B, P_P5_BE, P_P5_G, P_P5_W
};
enum { // net block, alphabetical
  N_C1_B = 0, N_C1_BE, N_C1_G, N_C1_W,
  N_CN_B, N_CN_BE, N_CN_G, N_CN_W,
  N_CP_B, N_CP_BE, N_CP_G, N_CP_W,
  N_CQ_B, N_CQ_W, N_CV_B, N_CV_W, N_CW_B, N_CW_W,
  N_N1_B, N_N1_W, N_N_B, N_N_W, N_NN_B, N_NN_W,
  N_PC1_B, N_PC1_BE, N_PC1_G, N_PC1_W,
  N_PC2_B, N_PC2_BE, N_PC2_G, N_PC2_W,
  N_PC3_B, N_PC3_BE, N_PC3_G, N_PC3_W
};

// ---------------------------------------------------------------------------
// Operand conversion: f32 -> zero-padded f16.
//   WH (Op,Cp)  row-major weights
//   XT (B,NP,Cp) activations, transposed to n-major so B fragments are
//                contiguous 16B-aligned runs.
// ---------------------------------------------------------------------------
__global__ void convert_w_kernel(const float* __restrict__ W, _Float16* __restrict__ WH,
                                 int O, int C, int Cp, long total)
{
  long i = (long)blockIdx.x * blockDim.x + threadIdx.x;
  if (i >= total) return;  // total = Op*Cp
  int r = (int)(i / Cp), c = (int)(i % Cp);
  WH[i] = (r < O && c < C) ? (_Float16)W[(size_t)r * C + c] : (_Float16)0.f;
}

__global__ void convert_xt_kernel(const float* __restrict__ X, _Float16* __restrict__ XT,
                                  int C, int NP, int Cp, long total)
{
  long i = (long)blockIdx.x * blockDim.x + threadIdx.x;
  if (i >= total) return;  // total = B*NP*Cp
  long per = (long)NP * Cp;
  long b = i / per; long r = i % per;
  int n = (int)(r / Cp), c = (int)(r % Cp);
  XT[i] = (c < C) ? (_Float16)X[((size_t)b * C + c) * NP + n] : (_Float16)0.f;
}

// ---------------------------------------------------------------------------
// Core GEMM:  Y[b,o,n] = sum_c W[o,c]*X[b,c,n] + bias[o]  (optional ReLU)
// Wave: one 16-row M tile x four 16-col N subtiles (A reuse, 4 indep WMMA
// chains covering the WMMA co-exec hazard window).
// Inner loop: 10 aligned b128 loads + 4 wmma, no guards, no conversions.
// A frag: a[j]=W[m][k0+kh+j], a[8+j]=W[m][k0+kh+16+j], kh = (lane<16)?0:8.
// B frag mirrored with n across lanes (from n-major XT).
// C/D: 8 f32 VGPRs, n=lane&15, rows = tm*16 + i (+8 for upper lanes).
// ---------------------------------------------------------------------------
__global__ void gemm_wmma_kernel(const _Float16* __restrict__ XT, // (B,NP,Cp)
                                 const _Float16* __restrict__ WH, // (Op,Cp)
                                 const float* __restrict__ bias,
                                 float* __restrict__ Y,           // (B,O,NP)
                                 int Cp, int O, int NP, int relu,
                                 int tiles, int tilesNQ)
{
  int b    = blockIdx.y;
  int wave = threadIdx.x >> 5;
  int lane = threadIdx.x & 31;
  int tile = blockIdx.x * 4 + wave;
  if (tile >= tiles) return;
  int tm = tile / tilesNQ, tq = tile % tilesNQ;
  int row     = tm * 16 + (lane & 15);
  int khalf   = (lane < 16) ? 0 : 8;
  int colbase = tq * 64;

  const _Float16* wr = WH + (size_t)row * Cp + khalf;
  const _Float16* xp[4];
  bool has[4];
#pragma unroll
  for (int j = 0; j < 4; ++j) {
    has[j] = (colbase + j * 16) < NP;             // wave-uniform
    int cj = has[j] ? (colbase + j * 16) : colbase;  // clamp to valid address
    xp[j] = XT + ((size_t)b * NP + cj + (lane & 15)) * Cp + khalf;
  }

  v8f acc[4];
  v8f zz = {0.f,0.f,0.f,0.f,0.f,0.f,0.f,0.f};
#pragma unroll
  for (int j = 0; j < 4; ++j) acc[j] = zz;

  for (int k0 = 0; k0 < Cp; k0 += 32) {
    if (k0 + 32 < Cp) {  // global_prefetch_b8: hide HBM latency for next slab
      __builtin_prefetch(wr + k0 + 32, 0, 3);
      __builtin_prefetch(xp[0] + k0 + 32, 0, 3);
      __builtin_prefetch(xp[2] + k0 + 32, 0, 3);
    }
    v8h alo = *(const v8h*)(wr + k0);
    v8h ahi = *(const v8h*)(wr + k0 + 16);
    v16h a  = __builtin_shufflevector(alo, ahi,
                                      0,1,2,3,4,5,6,7,8,9,10,11,12,13,14,15);
#pragma unroll
    for (int j = 0; j < 4; ++j) {
      v8h bl = *(const v8h*)(xp[j] + k0);
      v8h bh = *(const v8h*)(xp[j] + k0 + 16);
      v16h bf = __builtin_shufflevector(bl, bh,
                                        0,1,2,3,4,5,6,7,8,9,10,11,12,13,14,15);
      acc[j] = __builtin_amdgcn_wmma_f32_16x16x32_f16(false, a, false, bf,
                                                      (short)0, acc[j],
                                                      false, false);
    }
  }
  int rbase = tm * 16 + ((lane < 16) ? 0 : 8);
  int cn    = lane & 15;
#pragma unroll
  for (int j = 0; j < 4; ++j) {
    if (!has[j]) continue;
    int cc = colbase + j * 16 + cn;
#pragma unroll
    for (int i = 0; i < 8; ++i) {
      int r = rbase + i;
      if (r < O) {
        float v = acc[j][i] + (bias ? bias[r] : 0.f);
        if (relu) v = fmaxf(v, 0.f);
        Y[((size_t)b * O + r) * NP + cc] = v;
      }
    }
  }
}

// ---------------------------------------------------------------------------
// BatchNorm (training batch stats): per-channel reduce + apply.
// ---------------------------------------------------------------------------
__global__ void bn_stats_kernel(const float* __restrict__ y, float* mean, float* inv,
                                int O, int N, int NP)
{
  int o = blockIdx.x;
  long per = (long)BB * N;
  float s = 0.f, q = 0.f;
  for (long i = threadIdx.x; i < per; i += blockDim.x) {
    long b = i / N; int n = (int)(i % N);
    float v = y[((size_t)b * O + o) * NP + n];
    s += v; q += v * v;
  }
  __shared__ float sh[256], sh2[256];
  sh[threadIdx.x] = s; sh2[threadIdx.x] = q; __syncthreads();
  for (int st = 128; st; st >>= 1) {
    if ((int)threadIdx.x < st) { sh[threadIdx.x] += sh[threadIdx.x + st];
                                 sh2[threadIdx.x] += sh2[threadIdx.x + st]; }
    __syncthreads();
  }
  if (threadIdx.x == 0) {
    float m = sh[0] / (float)per;
    float v = sh2[0] / (float)per - m * m;
    mean[o] = m; inv[o] = rsqrtf(fmaxf(v, 0.f) + 1e-5f);
  }
}

__global__ void bn_apply_kernel(float* y, const float* mean, const float* inv,
                                const float* g, const float* be,
                                int O, int N, int NP, int relu, long total)
{
  long i = (long)blockIdx.x * blockDim.x + threadIdx.x;
  if (i >= total) return;  // total = B*O*N (logical)
  long per = (long)O * N;
  long b = i / per; long r = i % per;
  int o = (int)(r / N), n = (int)(r % N);
  size_t idx = ((size_t)b * O + o) * NP + n;
  float v = (y[idx] - mean[o]) * inv[o] * g[o] + be[o];
  y[idx] = relu ? fmaxf(v, 0.f) : v;
}

// ---------------------------------------------------------------------------
// Misc elementwise / data-movement kernels (NP = storage stride, N = logical)
// ---------------------------------------------------------------------------
__global__ void zero_kernel(float* p, long total) {
  long i = (long)blockIdx.x * blockDim.x + threadIdx.x;
  if (i < total) p[i] = 0.f;
}

__global__ void xpose_kernel(const float* __restrict__ pcl, float* __restrict__ pos,
                             int N, int NP, long total) { // (B,N,3) -> (B,3,NP)
  long i = (long)blockIdx.x * blockDim.x + threadIdx.x;
  if (i >= total) return;
  long b = i / N; int n = (int)(i % N);
  const float* p = pcl + i * 3;
  float* o = pos + (size_t)b * 3 * NP;
  o[n] = p[0]; o[NP + n] = p[1]; o[2 * NP + n] = p[2];
}

__global__ void knn16_kernel(const float* __restrict__ pts, int* __restrict__ idx, int N)
{
  int b = blockIdx.y;
  int n = blockIdx.x * blockDim.x + threadIdx.x;
  if (n >= N) return;
  const float* p = pts + (size_t)b * N * 3;
  float qx = p[n * 3], qy = p[n * 3 + 1], qz = p[n * 3 + 2];
  const int KK = 17;
  float bd[KK]; int bi[KK];
  for (int i = 0; i < KK; ++i) { bd[i] = 3.4e38f; bi[i] = 0; }
  for (int m = 0; m < N; ++m) {
    float dx = p[m * 3] - qx, dy = p[m * 3 + 1] - qy, dz = p[m * 3 + 2] - qz;
    float d2 = dx * dx + dy * dy + dz * dz;
    if (d2 < bd[KK - 1]) {
      int j = KK - 1;
      while (j > 0 && bd[j - 1] > d2) { bd[j] = bd[j - 1]; bi[j] = bi[j - 1]; --j; }
      bd[j] = d2; bi[j] = m;
    }
  }
  for (int k = 0; k < 16; ++k) idx[((size_t)b * N + n) * 16 + k] = bi[k + 1];
}

__global__ void iota_kernel(int* idx, int N, long total) { // identity knn (K=1)
  long i = (long)blockIdx.x * blockDim.x + threadIdx.x;
  if (i < total) idx[i] = (int)(i % N);
}

// ---------------------------------------------------------------------------
// Edge conv (graph conv), recompute flavor: stats pass + apply/max pass.
// ---------------------------------------------------------------------------
__global__ void edge_stats_kernel(const float* __restrict__ x, const int* __restrict__ idx,
                                  const float* __restrict__ W, const float* __restrict__ bias,
                                  float* gsum, float* gsq, int C, int N, int NP, int K)
{
  int b = blockIdx.y;
  int n = blockIdx.x * blockDim.x + threadIdx.x;
  __shared__ float ssum[EDGE_O], ssq[EDGE_O];
  if (threadIdx.x < EDGE_O) { ssum[threadIdx.x] = 0.f; ssq[threadIdx.x] = 0.f; }
  __syncthreads();
  if (n < N) {
    const float* xb = x + (size_t)b * C * NP;
    float xi[48], xd[48];
    for (int c = 0; c < C; ++c) xi[c] = xb[(size_t)c * NP + n];
    float ls[EDGE_O], lq[EDGE_O];
    for (int o = 0; o < EDGE_O; ++o) { ls[o] = 0.f; lq[o] = 0.f; }
    for (int k = 0; k < K; ++k) {
      int j = idx[((size_t)b * N + n) * K + k];
      for (int c = 0; c < C; ++c) xd[c] = xb[(size_t)c * NP + j] - xi[c];
      for (int o = 0; o < EDGE_O; ++o) {
        const float* wr = W + (size_t)o * 2 * C;
        float acc = bias[o];
        for (int c = 0; c < C; ++c) acc += wr[c] * xi[c] + wr[C + c] * xd[c];
        ls[o] += acc; lq[o] += acc * acc;
      }
    }
    for (int o = 0; o < EDGE_O; ++o) { atomicAdd(&ssum[o], ls[o]); atomicAdd(&ssq[o], lq[o]); }
  }
  __syncthreads();
  if (threadIdx.x < EDGE_O) {
    atomicAdd(&gsum[threadIdx.x], ssum[threadIdx.x]);
    atomicAdd(&gsq[threadIdx.x],  ssq[threadIdx.x]);
  }
}

__global__ void finalize_stats_kernel(const float* gsum, const float* gsq,
                                      float* mean, float* inv, int O, float cnt)
{
  int o = blockIdx.x * blockDim.x + threadIdx.x;
  if (o >= O) return;
  float m = gsum[o] / cnt;
  float v = gsq[o] / cnt - m * m;
  mean[o] = m; inv[o] = rsqrtf(fmaxf(v, 0.f) + 1e-5f);
}

__global__ void edge_apply_kernel(const float* __restrict__ x, const int* __restrict__ idx,
                                  const float* __restrict__ W, const float* __restrict__ bias,
                                  const float* mean, const float* inv,
                                  const float* gg, const float* bb,
                                  float* __restrict__ out, int C, int N, int NP, int K)
{
  int b = blockIdx.y;
  int n = blockIdx.x * blockDim.x + threadIdx.x;
  if (n >= N) return;
  const float* xb = x + (size_t)b * C * NP;
  float xi[48], xd[48];
  for (int c = 0; c < C; ++c) xi[c] = xb[(size_t)c * NP + n];
  float mx[EDGE_O];
  for (int o = 0; o < EDGE_O; ++o) mx[o] = 0.f;  // relu'd values are >= 0
  for (int k = 0; k < K; ++k) {
    int j = idx[((size_t)b * N + n) * K + k];
    for (int c = 0; c < C; ++c) xd[c] = xb[(size_t)c * NP + j] - xi[c];
    for (int o = 0; o < EDGE_O; ++o) {
      const float* wr = W + (size_t)o * 2 * C;
      float acc = bias[o];
      for (int c = 0; c < C; ++c) acc += wr[c] * xi[c] + wr[C + c] * xd[c];
      float z = (acc - mean[o]) * inv[o] * gg[o] + bb[o];
      mx[o] = fmaxf(mx[o], fmaxf(z, 0.f));
    }
  }
  for (int o = 0; o < EDGE_O; ++o) out[((size_t)b * EDGE_O + o) * NP + n] = mx[o];
}

// ---------------------------------------------------------------------------
// Distance weights, pooling, concat helpers
// ---------------------------------------------------------------------------
__global__ void dist_weight_kernel(const float* __restrict__ pos, const float* alpha,
                                   const float* beta, float* dw, int N, int NP)
{
  int b = blockIdx.x;
  __shared__ float sh[256];
  const float* p = pos + (size_t)b * 3 * NP;
  float a = alpha[0], bt = beta[0];
  float loc = 0.f;
  for (int n = threadIdx.x; n < N; n += 256) {
    float x = p[n], y = p[NP + n], z = p[2 * NP + n];
    float d = sqrtf(x * x + y * y + z * z);
    float w = 1.f / (1.f + __expf(a * d - bt));
    dw[(size_t)b * NP + n] = w;
    loc += w;
  }
  sh[threadIdx.x] = loc; __syncthreads();
  for (int st = 128; st; st >>= 1) {
    if ((int)threadIdx.x < st) sh[threadIdx.x] += sh[threadIdx.x + st];
    __syncthreads();
  }
  float s = sh[0];
  float den = s + ((s == 0.f) ? 1.f : 0.f) + 1e-6f;
  float scl = (float)N / den;
  for (int n = threadIdx.x; n < N; n += 256) dw[(size_t)b * NP + n] *= scl;
}

__global__ void wmean_kernel(const float* __restrict__ x, const float* __restrict__ dw,
                             float* g, int C, int Nin, int NPin, int dwStride, long total)
{
  long i = (long)blockIdx.x * blockDim.x + threadIdx.x;
  if (i >= total) return;  // total = B*C
  long b = i / C; int c = (int)(i % C);
  const float* xp = x + ((size_t)b * C + c) * NPin;
  const float* wp = dw + (size_t)b * dwStride;
  float s = 0.f;
  for (int n = 0; n < Nin; ++n) s += xp[n] * wp[n];
  g[i] = s / (float)Nin;
}

__global__ void pile_cat_kernel(const float* __restrict__ x, const float* __restrict__ g,
                                float* __restrict__ dst, int C, int NPin,
                                int Nout, int NPout, long total)
{
  long i = (long)blockIdx.x * blockDim.x + threadIdx.x;
  if (i >= total) return;  // total = B*2C*Nout
  long per = (long)2 * C * Nout;
  long b = i / per; long r = i % per;
  int ch = (int)(r / Nout); int n = (int)(r % Nout);
  float v = (ch < C) ? x[((size_t)b * C + ch) * NPin + n]
                     : g[(size_t)b * C + (ch - C)];
  dst[((size_t)b * 2 * C + ch) * NPout + n] = v;
}

__global__ void chan_copy_kernel(float* __restrict__ dst, const float* __restrict__ src,
                                 int Cd, int chOff, int Cs, int N, int NP, long total)
{
  long i = (long)blockIdx.x * blockDim.x + threadIdx.x;
  if (i >= total) return;  // total = B*Cs*N
  long per = (long)Cs * N;
  long b = i / per; long r = i % per;
  int c = (int)(r / N); int n = (int)(r % N);
  dst[((size_t)b * Cd + chOff + c) * NP + n] = src[((size_t)b * Cs + c) * NP + n];
}

__global__ void add_trunc_kernel(float* __restrict__ dst, const float* __restrict__ src,
                                 int C, int M, int NPd, int NPs, long total)
{
  long i = (long)blockIdx.x * blockDim.x + threadIdx.x;
  if (i >= total) return;  // total = B*C*M
  long per = (long)C * M;
  long b = i / per; long r = i % per;
  int c = (int)(r / M); int n = (int)(r % M);
  dst[((size_t)b * C + c) * NPd + n] += src[((size_t)b * C + c) * NPs + n];
}

__global__ void colmax_kernel(const float* __restrict__ x, float* out,
                              int C, int N, int NP, long total)
{
  long i = (long)blockIdx.x * blockDim.x + threadIdx.x;
  if (i >= total) return;  // total = B*C
  long b = i / C; int c = (int)(i % C);
  const float* xp = x + ((size_t)b * C + c) * NP;
  float m = xp[0];
  for (int n = 1; n < N; ++n) m = fmaxf(m, xp[n]);
  out[i] = m;
}

__global__ void truncrows_kernel(float* dst, const float* src, int M, int NPsrc, long total)
{
  long i = (long)blockIdx.x * blockDim.x + threadIdx.x;
  if (i >= total) return;  // total = B*M, dst compact stride M
  long b = i / M; int n = (int)(i % M);
  dst[i] = src[(size_t)b * NPsrc + n];
}

__global__ void wsig_kernel(const float* z, float* wbuf, float* outw,
                            int N, int NPz, long total)
{
  long i = (long)blockIdx.x * blockDim.x + threadIdx.x;
  if (i >= total) return;  // total = B*N
  long b = i / N; int n = (int)(i % N);
  float v = 0.01f + 1.f / (1.f + __expf(-z[(size_t)b * NPz + n]));
  wbuf[i] = v; outw[i] = v;
}

__global__ void mulw_kernel(float* dst, const float* x, const float* wgt,
                            int C, int N, int NP, long total)
{
  long i = (long)blockIdx.x * blockDim.x + threadIdx.x;
  if (i >= total) return;  // total = B*C*N, wgt compact (B,N)
  long per = (long)C * N;
  long b = i / per; long r = i % per;
  int c = (int)(r / N); int n = (int)(r % N);
  size_t idx = ((size_t)b * C + c) * NP + n;
  dst[idx] = x[idx] * wgt[(size_t)b * N + n];
}

// attn = max_c softmax_n(q);  fw[b,c] = sum_n v[b,c,n]*attn[b,n]
__global__ void attn_fw_kernel(const float* __restrict__ q, const float* __restrict__ v,
                               float* fw, int N, int NP)
{
  int b = blockIdx.x;
  int c = threadIdx.x;  // blockDim = 64
  __shared__ float pmat[64 * 48];
  __shared__ float am[48];
  const float* qr = q + ((size_t)b * 64 + c) * NP;
  float mx = -3.4e38f;
  for (int n = 0; n < N; ++n) mx = fmaxf(mx, qr[n]);
  float e[48]; float s = 0.f;
  for (int n = 0; n < N; ++n) { e[n] = __expf(qr[n] - mx); s += e[n]; }
  for (int n = 0; n < N; ++n) pmat[c * 48 + n] = e[n] / s;
  __syncthreads();
  if (c < N) {
    float m = 0.f;
    for (int cc = 0; cc < 64; ++cc) m = fmaxf(m, pmat[cc * 48 + c]);
    am[c] = m;
  }
  __syncthreads();
  const float* vr = v + ((size_t)b * 64 + c) * NP;
  float acc = 0.f;
  for (int n = 0; n < N; ++n) acc += vr[n] * am[n];
  fw[(size_t)b * 64 + c] = acc;
}

__global__ void head_kernel(const float* fw, const float* n1w, const float* n1b,
                            const float* nw, const float* nb, float* out, int Bn)
{
  int b = blockIdx.x * blockDim.x + threadIdx.x;
  if (b >= Bn) return;
  const float* f = fw + (size_t)b * 64;
  float h[64];
  for (int o = 0; o < 64; ++o) {
    float a = n1b[o];
    for (int c = 0; c < 64; ++c) a += n1w[(size_t)o * 64 + c] * f[c];
    h[o] = a;
  }
  float n4[4];
  for (int o = 0; o < 4; ++o) {
    float a = nb[o];
    for (int c = 0; c < 64; ++c) a += nw[(size_t)o * 64 + c] * h[c];
    n4[o] = a;
  }
  float nn = sqrtf(n4[0] * n4[0] + n4[1] * n4[1] + n4[2] * n4[2]);
  float d = fmaxf(nn, 1e-12f);
  out[b * 3 + 0] = n4[0] / d;
  out[b * 3 + 1] = n4[1] / d;
  out[b * 3 + 2] = n4[2] / d;
}

__global__ void neigh_kernel(const float* __restrict__ feat, const float* w,
                             const float* bias, float* out, int N, int NP, long total)
{
  long i = (long)blockIdx.x * blockDim.x + threadIdx.x;
  if (i >= total) return;  // total = B*N
  long b = i / N; int n = (int)(i % N);
  float a[3];
  for (int o = 0; o < 3; ++o) {
    float s = bias[o];
    for (int c = 0; c < 128; ++c)
      s += feat[((size_t)b * 128 + c) * NP + n] * w[(size_t)o * 128 + c];
    a[o] = s;
  }
  float nn = sqrtf(a[0] * a[0] + a[1] * a[1] + a[2] * a[2]);
  float d = fmaxf(nn, 1e-12f);
  out[i * 3 + 0] = a[0] / d;
  out[i * 3 + 1] = a[1] / d;
  out[i * 3 + 2] = a[2] / d;
}

// ---------------------------------------------------------------------------
// Host-side orchestration helpers
// ---------------------------------------------------------------------------
struct WS {
  float *POS, *CODE, *CODE2, *POSF, *T0, *T1, *YY, *CAT, *Y, *DW, *G;
  float *MEAN, *INV, *ESUM, *ESQ;
  _Float16 *XT, *WH;
  int *IDX;
};

#define EW(kern, total, ...) do { \
  long _t = (total); \
  kern<<<(unsigned)cdivl(_t, 256), 256, 0, s>>>(__VA_ARGS__, _t); \
} while (0)

static void gemm(hipStream_t s, WS& w, const float* X, const float* W, const float* bias,
                 float* Y, int C, int O, int N, int relu)
{
  int NP = ALN16(N), Cp = ALN32(C), Op = ALN16(O);
  EW(convert_w_kernel,  (long)Op * Cp, W, w.WH, O, C, Cp);
  EW(convert_xt_kernel, (long)BB * NP * Cp, X, w.XT, C, NP, Cp);
  int tilesNQ = (NP + 63) / 64;
  int tiles = (Op / 16) * tilesNQ;
  dim3 grid((unsigned)((tiles + 3) / 4), BB);
  gemm_wmma_kernel<<<grid, 128, 0, s>>>(w.XT, w.WH, bias, Y, Cp, O, NP,
                                        relu, tiles, tilesNQ);
}

static void bn(hipStream_t s, WS& w, float* Y, const float* g, const float* be,
               int O, int N, int NP, int relu)
{
  bn_stats_kernel<<<O, 256, 0, s>>>(Y, w.MEAN, w.INV, O, N, NP);
  EW(bn_apply_kernel, (long)BB * O * N, Y, w.MEAN, w.INV, g, be, O, N, NP, relu);
}

static void pile(hipStream_t s, WS& w, const float* x, int C, int Nin, int NPin,
                 int Nout, int NPout, const float* dw, int dwStride,
                 const float* Wt, const float* bias, const float* g, const float* be,
                 float* out, int O)
{
  EW(wmean_kernel, (long)BB * C, x, dw, w.G, C, Nin, NPin, dwStride);
  EW(pile_cat_kernel, (long)BB * 2 * C * Nout, x, w.G, w.CAT, C, NPin, Nout, NPout);
  gemm(s, w, w.CAT, Wt, bias, out, 2 * C, O, Nout, 0);
  bn(s, w, out, g, be, O, Nout, NPout, 1);
}

static void edge_conv(hipStream_t s, WS& w, const float* x, const int* idx,
                      const float* Wt, const float* bias, const float* g,
                      const float* be, float* out, int C, int N, int NP, int K)
{
  EW(zero_kernel, 64L, w.ESUM);
  EW(zero_kernel, 64L, w.ESQ);
  dim3 grid((unsigned)cdivl(N, 128), BB);
  edge_stats_kernel<<<grid, 128, 0, s>>>(x, idx, Wt, bias, w.ESUM, w.ESQ, C, N, NP, K);
  finalize_stats_kernel<<<1, EDGE_O, 0, s>>>(w.ESUM, w.ESQ, w.MEAN, w.INV,
                                             EDGE_O, (float)((long)BB * N * K));
  edge_apply_kernel<<<grid, 128, 0, s>>>(x, idx, Wt, bias, w.MEAN, w.INV,
                                         g, be, out, C, N, NP, K);
}

#define PF(e) ((const float*)P[e])

static void encoder(hipStream_t s, void* const* P, const float* pcl,
                    int N, int npcl, int K, WS& w, float* outY, float* outWD)
{
  int NP = ALN16(N);
  int N1 = 2 * npcl, NP1 = ALN16(N1);
  int N2 = npcl,     NP2 = ALN16(N2);
  int NH = npcl / 2, NPH = ALN16(NH);
  // pos = transpose(pcl)
  EW(xpose_kernel, (long)BB * N, pcl, w.POS, N, NP);
  // knn indices
  if (K == 16) {
    dim3 g((unsigned)cdivl(N, 128), BB);
    knn16_kernel<<<g, 128, 0, s>>>(pcl, w.IDX, N);
  } else {
    EW(iota_kernel, (long)BB * N, w.IDX, N);
  }
  // graph convs g0 (3 -> 24), g1 (24 -> 24)
  edge_conv(s, w, w.POS, w.IDX, PF(P_G0_W), PF(P_G0_B), PF(P_G0_G), PF(P_G0_BE),
            w.CODE, 3, N, NP, K);
  edge_conv(s, w, w.CODE, w.IDX, PF(P_G1_W), PF(P_G1_B), PF(P_G1_G), PF(P_G1_BE),
            w.CODE2, 24, N, NP, K);
  // fusion net
  gemm(s, w, w.POS, PF(P_M10_W), PF(P_M10_B), w.T0, 3, 64, N, 1);
  gemm(s, w, w.T0,  PF(P_M11_W), PF(P_M11_B), w.POSF, 64, 64, N, 0);
  EW(chan_copy_kernel, (long)BB * 64 * N, w.CAT, w.POSF, 88, 0, 64, N, NP);
  EW(chan_copy_kernel, (long)BB * 24 * N, w.CAT, w.CODE2, 88, 64, 24, N, NP);
  gemm(s, w, w.CAT, PF(P_LIN0_W), PF(P_LIN0_B), w.T0, 88, 128, N, 1);
  gemm(s, w, w.T0,  PF(P_LIN1_W), PF(P_LIN1_B), w.T1, 128, 128, N, 1);
  gemm(s, w, w.T1,  PF(P_LIN2_W), PF(P_LIN2_B), w.T0, 128, 40, N, 1);
  EW(chan_copy_kernel, (long)BB * 40 * N, w.CAT, w.T0, 128, 0, 40, N, NP);
  EW(chan_copy_kernel, (long)BB * 64 * N, w.CAT, w.POSF, 128, 40, 64, N, NP);
  EW(chan_copy_kernel, (long)BB * 24 * N, w.CAT, w.CODE2, 128, 104, 24, N, NP);
  gemm(s, w, w.CAT, PF(P_LIN3_W), PF(P_LIN3_B), w.T1, 128, 128, N, 1);
  gemm(s, w, w.T1,  PF(P_LIN4_W), PF(P_LIN4_B), w.YY, 128, 64, N, 0);
  // c1: concat(code, yy) -> 128 + BN + ReLU
  EW(chan_copy_kernel, (long)BB * 24 * N, w.CAT, w.CODE2, 88, 0, 24, N, NP);
  EW(chan_copy_kernel, (long)BB * 64 * N, w.CAT, w.YY, 88, 24, 64, N, NP);
  gemm(s, w, w.CAT, PF(P_C1_W), PF(P_C1_B), w.Y, 88, 128, N, 0);
  bn(s, w, w.Y, PF(P_C1_G), PF(P_C1_BE), 128, N, NP, 1);
  // distance weights
  dist_weight_kernel<<<BB, 256, 0, s>>>(w.POS, PF(P_ALPHA), PF(P_BETA), w.DW, N, NP);
  // pile convs
  pile(s, w, w.Y, 128, N, NP, N1, NP1, w.DW, NP,
       PF(P_P1_W), PF(P_P1_B), PF(P_P1_G), PF(P_P1_BE), w.T0, 256);
  pile(s, w, w.T0, 256, N1, NP1, N2, NP2, w.DW, NP,
       PF(P_P2_W), PF(P_P2_B), PF(P_P2_G), PF(P_P2_BE), w.T1, 256);
  EW(add_trunc_kernel, (long)BB * 256 * N2, w.T1, w.T0, 256, N2, NP2, NP1);
  pile(s, w, w.T1, 256, N2, NP2, N2, NP2, w.DW, NP,
       PF(P_P3_W), PF(P_P3_B), PF(P_P3_G), PF(P_P3_BE), w.T0, 256);
  pile(s, w, w.T0, 256, N2, NP2, NH, NPH, w.DW, NP,
       PF(P_P4_W), PF(P_P4_B), PF(P_P4_G), PF(P_P4_BE), w.T1, 256);
  EW(add_trunc_kernel, (long)BB * 256 * NH, w.T1, w.T0, 256, NH, NPH, NP2);
  pile(s, w, w.T1, 256, NH, NPH, NH, NPH, w.DW, NP,
       PF(P_P5_W), PF(P_P5_B), PF(P_P5_G), PF(P_P5_BE), w.T0, 256);
  // c2 / c3
  gemm(s, w, w.T0, PF(P_C2_W), PF(P_C2_B), w.T1, 256, 256, NH, 0);
  bn(s, w, w.T1, PF(P_C2_G), PF(P_C2_BE), 256, NH, NPH, 1);
  gemm(s, w, w.T1, PF(P_C3_W), PF(P_C3_B), outY, 256, 128, NH, 0);
  bn(s, w, outY, PF(P_C3_G), PF(P_C3_BE), 128, NH, NPH, 1);
  if (outWD) EW(truncrows_kernel, (long)BB * NH, outWD, w.DW, NH, NP);
}

// ---------------------------------------------------------------------------
// Entry point
// ---------------------------------------------------------------------------
extern "C" void kernel_launch(void* const* d_in, const int* in_sizes, int n_in,
                              void* d_out, int out_size, void* d_ws, size_t ws_size,
                              hipStream_t stream)
{
  (void)in_sizes; (void)n_in; (void)out_size; (void)ws_size;
  hipStream_t s = stream;
  const float* pcl_pat = (const float*)d_in[0];
  const float* pcl_sample = (const float*)d_in[1];
  void* const* NETP = d_in + 2;        // 36 leaves
  void* const* PEP  = d_in + 2 + 36;   // 56 leaves
  void* const* PGP  = d_in + 2 + 36 + 56;

  // workspace arena (bump allocator, deterministic each call)
  char* base = (char*)d_ws;
  size_t off = 0;
  auto AB = [&](size_t nbytes) -> void* {
    void* p = (void*)(base + off);
    off += (nbytes + 255) & ~(size_t)255;
    return p;
  };
  auto AF = [&](size_t nfl) -> float* { return (float*)AB(nfl * 4); };
  auto AH = [&](size_t nh)  -> _Float16* { return (_Float16*)AB(nh * 2); };

  WS w;
  w.POS   = AF((size_t)BB * 3 * 1216);
  w.IDX   = (int*)AB((size_t)BB * NPAT * 16 * 4);
  w.CODE  = AF((size_t)BB * 24 * 1216);
  w.CODE2 = AF((size_t)BB * 24 * 1216);
  w.POSF  = AF((size_t)BB * 64 * 1216);
  w.T0    = AF((size_t)BB * 163840);   // max(128x1200, 256x608, 512x304)
  w.T1    = AF((size_t)BB * 163840);
  w.YY    = AF((size_t)BB * 64 * 1216);
  w.CAT   = AF((size_t)BB * 163840);
  w.Y     = AF((size_t)BB * 128 * 1216);
  w.DW    = AF((size_t)BB * 1216);
  w.G     = AF((size_t)BB * 256);
  w.MEAN  = AF(512); w.INV = AF(512);
  w.ESUM  = AF(64);  w.ESQ = AF(64);
  w.XT    = AH((size_t)BB * 163840);   // f16 transposed activations
  w.WH    = AH((size_t)256 * 512);     // f16 padded weights
  float* PE_Y  = AF((size_t)BB * 128 * 96);
  float* PE_WD = AF((size_t)BB * 87);
  float* PG_Y  = AF((size_t)BB * 128 * 160);
  float* YGMAX = AF((size_t)BB * 128);
  float* Y0 = AF((size_t)BB * 128 * 96);
  float* Y1 = AF((size_t)BB * 128 * 96);
  float* Y2 = AF((size_t)BB * 128 * 96);
  float* FEAT0 = AF((size_t)BB * 128 * 48);
  float* FEAT  = AF((size_t)BB * 128 * 48);
  float* FEATW = AF((size_t)BB * 128 * 48);
  float* TMP   = AF((size_t)BB * 128 * 48);
  float* ZB    = AF((size_t)BB * 48);
  float* WBUF  = AF((size_t)BB * 43);
  float* Q = AF((size_t)BB * 64 * 48);
  float* V = AF((size_t)BB * 64 * 48);
  float* FWV = AF((size_t)BB * 64);

  // ---- patch encoder (N=700, knn=16) and global encoder (N=1200, knn=1) ----
  encoder(s, PEP, pcl_pat, NPAT, 175, 16, w, PE_Y, PE_WD);
  encoder(s, PGP, pcl_sample, NSAM, 300, 1, w, PG_Y, nullptr);
  EW(colmax_kernel, (long)BB * 128, PG_Y, YGMAX, 128, 150, 160);

  // ---- decoder / head (N=87 stride 96, N=43 stride 48) ----
  void* const* P = NETP;
  EW(pile_cat_kernel, (long)BB * 256 * 87, PE_Y, YGMAX, w.CAT, 128, 96, 87, 96);
  gemm(s, w, w.CAT, PF(N_CP_W), PF(N_CP_B), Y0, 256, 128, 87, 0);
  bn(s, w, Y0, PF(N_CP_G), PF(N_CP_BE), 128, 87, 96, 1);
  pile(s, w, Y0, 128, 87, 96, 87, 96, PE_WD, 87,
       PF(N_PC1_W), PF(N_PC1_B), PF(N_PC1_G), PF(N_PC1_BE), Y1, 128);
  pile(s, w, Y1, 128, 87, 96, 43, 48, PE_WD, 87,
       PF(N_PC2_W), PF(N_PC2_B), PF(N_PC2_G), PF(N_PC2_BE), Y2, 128);
  EW(add_trunc_kernel, (long)BB * 128 * 43, Y2, Y1, 128, 43, 48, 96);
  EW(add_trunc_kernel, (long)BB * 128 * 43, Y2, Y0, 128, 43, 48, 96);
  pile(s, w, Y2, 128, 43, 48, 43, 48, PE_WD, 87,
       PF(N_PC3_W), PF(N_PC3_B), PF(N_PC3_G), PF(N_PC3_BE), FEAT0, 128);
  gemm(s, w, FEAT0, PF(N_C1_W), PF(N_C1_B), FEAT, 128, 128, 43, 0);
  bn(s, w, FEAT, PF(N_C1_G), PF(N_C1_BE), 128, 43, 48, 1);
  // weights = 0.01 + sigmoid(cw(feat))  (also 2nd output)
  float* out_f = (float*)d_out;
  gemm(s, w, FEAT, PF(N_CW_W), PF(N_CW_B), ZB, 128, 1, 43, 0);
  EW(wsig_kernel, (long)BB * 43, ZB, WBUF, out_f + (size_t)BB * 3, 43, 48);
  // feat_w = cbr(feat * weights, 'cn')
  EW(mulw_kernel, (long)BB * 128 * 43, TMP, FEAT, WBUF, 128, 43, 48);
  gemm(s, w, TMP, PF(N_CN_W), PF(N_CN_B), FEATW, 128, 128, 43, 0);
  bn(s, w, FEATW, PF(N_CN_G), PF(N_CN_BE), 128, 43, 48, 1);
  // q / v, attention-weighted feature, final MLP -> normal
  gemm(s, w, FEATW, PF(N_CQ_W), PF(N_CQ_B), Q, 128, 64, 43, 0);
  gemm(s, w, FEATW, PF(N_CV_W), PF(N_CV_B), V, 128, 64, 43, 0);
  attn_fw_kernel<<<BB, 64, 0, s>>>(Q, V, FWV, 43, 48);
  head_kernel<<<1, BB, 0, s>>>(FWV, PF(N_N1_W), PF(N_N1_B),
                               PF(N_N_W), PF(N_N_B), out_f, BB);
  // neigh = l2norm(feat^T @ nn_w^T + b)  (3rd output)
  EW(neigh_kernel, (long)BB * 43, FEAT, PF(N_NN_W), PF(N_NN_B),
     out_f + (size_t)BB * 3 + (size_t)BB * 43, 43, 48);
}
